// NeuralREGCOLING_17927193494448
// MI455X (gfx1250) — compile-verified, matrix operand order checked by reading, not run on hardware
//
#include <hip/hip_runtime.h>
#include <hip/hip_bf16.h>
#include <math.h>

// ---------------------------------------------------------------------------
// Types for CDNA5 WMMA (wave32): V_WMMA_F32_16X16X32_BF16
// ---------------------------------------------------------------------------
typedef __bf16 bf16_t;
typedef __attribute__((ext_vector_type(16))) __bf16 v16bf;
typedef __attribute__((ext_vector_type(8)))  __bf16 v8bf;
typedef __attribute__((ext_vector_type(8)))  float  v8f;
typedef __attribute__((ext_vector_type(4)))  unsigned int u32x4;
typedef __attribute__((ext_vector_type(8)))  int i32x8;
typedef __attribute__((ext_vector_type(4)))  int i32x4;

union FragBF { v16bf v; v8bf h[2]; };

// Fragment load: per lane, elements 0..7 are K = kbase..kbase+7 and elements
// 8..15 are K = kbase+16..kbase+23 (kbase = (lane>>4)*8), i.e. two 16B loads
// from a K-contiguous row.  Works for both LDS and global pointers.
__device__ __forceinline__ v16bf load_frag(const bf16_t* p) {
  FragBF f;
  f.h[0] = *(const v8bf*)(p);
  f.h[1] = *(const v8bf*)(p + 16);
  return f.v;
}

__device__ __forceinline__ v8f wmma_bf16(v16bf a, v16bf b, v8f c) {
  return __builtin_amdgcn_wmma_f32_16x16x32_bf16(false, a, false, b, (short)0, c, false, false);
}

__device__ __forceinline__ float sigm(float x) { return 1.0f / (1.0f + __expf(-x)); }

// ---------------------------------------------------------------------------
// TDM: DMA a 2D tile (rows x K bf16, row stride = K elements) from global to
// LDS.  D# built per CDNA5 ISA (group0 128b, group1 256b).
// Issued once per block by wave 0; completion via s_wait_tensorcnt.
// ---------------------------------------------------------------------------
__device__ __forceinline__ void tdm_load_tile_bf16(const bf16_t* gsrc, void* lds_dst,
                                                   int rows, int K) {
  unsigned long long ga = (unsigned long long)gsrc;
  unsigned lds_addr = (unsigned)(unsigned long long)lds_dst;  // low 32 bits = LDS byte offset
  u32x4 g0;
  g0[0] = 1u;                                        // count=1, user descriptor
  g0[1] = lds_addr;                                  // lds_addr [63:32]
  g0[2] = (unsigned)(ga & 0xffffffffu);              // global_addr [95:64]
  g0[3] = (unsigned)((ga >> 32) & 0x01ffffffu)       // global_addr [120:96]
        | (2u << 30);                                // type=2 ("image")
  i32x8 g1;
  g1[0] = (1 << 16);                                 // data_size=1 (2 bytes/elem)
  g1[1] = (K & 0xffff) << 16;                        // tensor_dim0 low16  (bits 63:48)
  g1[2] = ((unsigned)K >> 16) | (rows << 16);        // tensor_dim0 hi | tensor_dim1 lo
  g1[3] = ((K & 0xffff) << 16);                      // tensor_dim1 hi(0) | tile_dim0=K
  g1[4] = rows;                                      // tile_dim1=rows, tile_dim2=0
  g1[5] = K;                                         // tensor_dim0_stride low32 (elems)
  g1[6] = 0;                                         // stride hi16 | dim1_stride lo16
  g1[7] = 0;
  i32x4 g2 = { 0, 0, 0, 0 };                         // 2D tensor: groups 2/3 unused
  i32x4 g3 = { 0, 0, 0, 0 };
#if defined(__clang_major__) && (__clang_major__ >= 23)
  __builtin_amdgcn_tensor_load_to_lds(g0, g1, g2, g3, g1, 0);
#else
  __builtin_amdgcn_tensor_load_to_lds(g0, g1, g2, g3, 0);
#endif
  __builtin_amdgcn_s_wait_tensorcnt(0);
}

// ---------------------------------------------------------------------------
// Utility kernels
// ---------------------------------------------------------------------------
__global__ void k_f32_to_bf16(const float* __restrict__ src, bf16_t* __restrict__ dst, int n) {
  int g = blockIdx.x * blockDim.x + threadIdx.x;
  if (g < n) dst[g] = (bf16_t)src[g];
}

__global__ void k_zero_f32(float* p, int n) {
  int g = blockIdx.x * blockDim.x + threadIdx.x;
  if (g < n) p[g] = 0.0f;
}

// Gather embedding rows -> bf16, with optional time reversal. out: (L*B, E)
__global__ void k_gather_bf16(const float* __restrict__ emb, const int* __restrict__ idx,
                              bf16_t* __restrict__ out, int L, int B, int E, int rev) {
  int g = blockIdx.x * blockDim.x + threadIdx.x;
  int total = L * B * E;
  if (g >= total) return;
  int row = g / E, c = g % E;
  int t = row / B, b = row % B;
  int ts = rev ? (L - 1 - t) : t;
  out[g] = (bf16_t)emb[(size_t)idx[ts * B + b] * E + c];
}

// Gather embedding rows -> f32 (refexes). out: (rows, E)
__global__ void k_gather_f32(const float* __restrict__ emb, const int* __restrict__ idx,
                             float* __restrict__ out, int rows, int E) {
  int g = blockIdx.x * blockDim.x + threadIdx.x;
  int total = rows * E;
  if (g >= total) return;
  int row = g / E, c = g % E;
  out[g] = emb[(size_t)idx[row] * E + c];
}

// enc[l] = [h_fwd[l], h_bwd[L-1-l]]  (f32 and bf16 copies)
__global__ void k_concat_enc(const float* __restrict__ hf, const float* __restrict__ hb,
                             float* __restrict__ enc, bf16_t* __restrict__ encb,
                             int L, int B, int H) {
  int g = blockIdx.x * blockDim.x + threadIdx.x;
  int twoH = 2 * H;
  int total = L * B * twoH;
  if (g >= total) return;
  int row = g / twoH, c = g % twoH;
  int l = row / B, b = row % B;
  float v = (c < H) ? hf[(size_t)row * H + c]
                    : hb[((size_t)(L - 1 - l) * B + b) * H + (c - H)];
  enc[g] = v;
  encb[g] = (bf16_t)v;
}

// ---------------------------------------------------------------------------
// Generic GEMM: C(MxN) = A(MxK,bf16) @ W(NxK,bf16)^T + bias(N)
// Block = 4 waves, owns a 16x256 output strip.  A tile (16xK) is DMA'd once
// into LDS via TDM; each wave computes 4 N-tiles, reusing the LDS A fragment
// across 4 WMMAs per K-chunk.  Requires M%16==0, N%256==0, K%32==0.
// ---------------------------------------------------------------------------
__global__ void k_gemm(const bf16_t* __restrict__ A, const bf16_t* __restrict__ W,
                       const float* __restrict__ bias, float* __restrict__ C,
                       int M, int N, int K) {
  extern __shared__ char smem[];
  bf16_t* shA = (bf16_t*)smem;                  // 16 x K bf16
  int nstrips = N >> 8;
  int tmi   = blockIdx.x / nstrips;
  int strip = blockIdx.x % nstrips;
  int wave = threadIdx.x >> 5;
  int lane = threadIdx.x & 31;
  int lrow = lane & 15;
  int lhi  = lane >> 4;

  if (wave == 0)                                // async-tensor DMA: A tile -> LDS
    tdm_load_tile_bf16(A + (size_t)tmi * 16 * K, shA, 16, K);
  __syncthreads();

  int n0 = strip * 256 + wave * 64;
  v8f acc[4];
  #pragma unroll
  for (int j = 0; j < 4; ++j)
    #pragma unroll
    for (int r = 0; r < 8; ++r) acc[j][r] = 0.0f;

  const bf16_t* arow = shA + (size_t)lrow * K + lhi * 8;
  for (int kk = 0; kk < K; kk += 32) {
    v16bf a = load_frag(arow + kk);             // ds_load_b128 x2, reused 4x
    #pragma unroll
    for (int j = 0; j < 4; ++j) {
      const bf16_t* bp = W + (size_t)(n0 + j * 16 + lrow) * K + kk + lhi * 8;
      __builtin_prefetch((const void*)(bp + 128), 0, 1);    // global_prefetch_b8
      acc[j] = wmma_bf16(a, load_frag(bp), acc[j]);
    }
  }

  #pragma unroll
  for (int j = 0; j < 4; ++j) {
    int n = n0 + j * 16 + lrow;
    float bv = bias ? bias[n] : 0.0f;
    #pragma unroll
    for (int r = 0; r < 8; ++r) {
      int m = tmi * 16 + r + lhi * 8;           // C layout: vgpr r -> M=r (+8 hi lanes)
      C[(size_t)m * N + n] = acc[j][r] + bv;
    }
  }
}

// ---------------------------------------------------------------------------
// Persistent LSTM kernel.  One workgroup (32 waves, 1024 thr) per run.
// gates(64,2048) = Gx[t] + h @ Whh^T + bhh ; i,f,g,o = split(gates)
// h kept double-buffered in LDS (bf16); c kept in registers (owner-stable).
// Wave w: tm = w>>3 (row tile of 16), jb = w&7 ; handles j-tiles jb+8q, q=0..3
// of the 512-wide per-gate space; for each j-tile computes all 4 gate tiles.
// ---------------------------------------------------------------------------
struct LstmDesc {
  const float*  Gx;       // (L*64, 2048), includes bih
  const bf16_t* Whh;      // (2048, 512) bf16
  const float*  bhh;      // (2048)
  float*        hs;       // (L*64, 512) f32 h output
  const float*  h0;       // (64,512) or null -> zeros
  const float*  c0;       // (64,512) or null -> zeros
  float*        c_out;    // (64,512) or null
  bf16_t*       hbf_out;  // (64,512) bf16 h of last step (decoder) or null
  int           L;
  int           pad;
};
struct LstmArgs { LstmDesc d[6]; };

__global__ __launch_bounds__(1024)
void k_lstm(LstmArgs args) {
  extern __shared__ char smem[];
  bf16_t* hbuf0 = (bf16_t*)smem;                // 64*512 bf16
  bf16_t* hbuf1 = hbuf0 + 64 * 512;

  const LstmDesc D = args.d[blockIdx.x];
  const int tid  = threadIdx.x;
  const int wave = tid >> 5;
  const int lane = tid & 31;
  const int tm   = wave >> 3;                   // 0..3
  const int jb   = wave & 7;                    // 0..7
  const int lrow = lane & 15;
  const int lhi  = lane >> 4;

  // init h (LDS) and c (registers)
  for (int i = tid; i < 64 * 512; i += 1024)
    hbuf0[i] = D.h0 ? (bf16_t)D.h0[i] : (bf16_t)0.0f;

  float creg[4][8];
  #pragma unroll
  for (int q = 0; q < 4; ++q) {
    int n = (jb + 8 * q) * 16 + lrow;
    #pragma unroll
    for (int r = 0; r < 8; ++r) {
      int m = tm * 16 + r + lhi * 8;
      creg[q][r] = D.c0 ? D.c0[m * 512 + n] : 0.0f;
    }
  }
  __syncthreads();

  bf16_t* hin  = hbuf0;
  bf16_t* hout = hbuf1;

  for (int t = 0; t < D.L; ++t) {
    const float* Gxt = D.Gx + (size_t)t * 64 * 2048;
    float*       hst = D.hs + (size_t)t * 64 * 512;

    for (int q = 0; q < 4; ++q) {
      int j0 = (jb + 8 * q) * 16;
      v8f acc[4];
      #pragma unroll
      for (int g = 0; g < 4; ++g)
        #pragma unroll
        for (int r = 0; r < 8; ++r) acc[g][r] = 0.0f;

      const bf16_t* arow = hin + (size_t)(tm * 16 + lrow) * 512 + lhi * 8;
      for (int kk = 0; kk < 512; kk += 32) {
        v16bf a = load_frag(arow + kk);         // ds_load_b128 x2
        #pragma unroll
        for (int g = 0; g < 4; ++g) {
          int nrow = g * 512 + j0 + lrow;       // Whh output row
          acc[g] = wmma_bf16(a, load_frag(D.Whh + (size_t)nrow * 512 + kk + lhi * 8), acc[g]);
        }
      }

      // elementwise cell update (writes go to the *other* h buffer)
      int n = j0 + lrow;
      float bi = D.bhh[n], bf = D.bhh[512 + n], bg = D.bhh[1024 + n], bo = D.bhh[1536 + n];
      #pragma unroll
      for (int r = 0; r < 8; ++r) {
        int m = tm * 16 + r + lhi * 8;
        const float* gx = Gxt + (size_t)m * 2048;
        float gi = acc[0][r] + gx[n]        + bi;
        float gf = acc[1][r] + gx[512 + n]  + bf;
        float gg = acc[2][r] + gx[1024 + n] + bg;
        float go = acc[3][r] + gx[1536 + n] + bo;
        float c = sigm(gf) * creg[q][r] + sigm(gi) * tanhf(gg);
        float h = sigm(go) * tanhf(c);
        creg[q][r] = c;
        hst[m * 512 + n]  = h;
        hout[m * 512 + n] = (bf16_t)h;
        if (D.hbf_out) D.hbf_out[m * 512 + n] = (bf16_t)h;
      }
    }
    __syncthreads();                            // step boundary
    bf16_t* tmp = hin; hin = hout; hout = tmp;
  }

  if (D.c_out) {
    #pragma unroll
    for (int q = 0; q < 4; ++q) {
      int n = (jb + 8 * q) * 16 + lrow;
      #pragma unroll
      for (int r = 0; r < 8; ++r) {
        int m = tm * 16 + r + lhi * 8;
        D.c_out[m * 512 + n] = creg[q][r];
      }
    }
  }
}

// ---------------------------------------------------------------------------
// Attention pieces
// ---------------------------------------------------------------------------
// score[l*B+b] = dot(tanh(e_enc[l,b,:] + sdec[b,:]), vW) + vb
__global__ void k_att_score(const float* __restrict__ eenc, const float* __restrict__ sdec,
                            const float* __restrict__ vW, const float* __restrict__ vb,
                            float* __restrict__ score, int B) {
  int lb = blockIdx.x;
  int tid = threadIdx.x;                        // 128
  int b = lb % B;
  const float* ee = eenc + (size_t)lb * 512;
  const float* sd = sdec + (size_t)b * 512;
  float acc = 0.0f;
  for (int j = tid; j < 512; j += 128) acc += tanhf(ee[j] + sd[j]) * vW[j];
  __shared__ float red[128];
  red[tid] = acc;
  for (int s = 64; s > 0; s >>= 1) { __syncthreads(); if (tid < s) red[tid] += red[tid + s]; }
  if (tid == 0) score[lb] = red[0] + vb[0];
}

// softmax over L + context reduction ctx[b,:] = sum_l alpha_l * enc[l,b,:]
__global__ void k_softmax_ctx(const float* __restrict__ score, const float* __restrict__ enc,
                              float* __restrict__ ctx, float* __restrict__ ae,
                              int L, int B, int twoH) {
  int b = blockIdx.x, tid = threadIdx.x;        // 256
  __shared__ float sh[128];
  __shared__ float red[256];
  float m = -1e30f;
  for (int l = tid; l < L; l += 256) { float v = score[l * B + b]; sh[l] = v; m = fmaxf(m, v); }
  red[tid] = m;
  for (int s = 128; s > 0; s >>= 1) { __syncthreads(); if (tid < s) red[tid] = fmaxf(red[tid], red[tid + s]); }
  __syncthreads(); m = red[0]; __syncthreads();
  float sum = 0.0f;
  for (int l = tid; l < L; l += 256) { float e = __expf(sh[l] - m); sh[l] = e; sum += e; }
  red[tid] = sum;
  for (int s = 128; s > 0; s >>= 1) { __syncthreads(); if (tid < s) red[tid] += red[tid + s]; }
  __syncthreads(); sum = red[0]; __syncthreads();
  float inv = 1.0f / sum;
  for (int l = tid; l < L; l += 256) { sh[l] *= inv; if (ae) ae[b * L + l] = sh[l]; }
  __syncthreads();
  for (int c = tid; c < twoH; c += 256) {
    float acc = 0.0f;
    for (int l = 0; l < L; ++l) acc += sh[l] * enc[((size_t)l * B + b) * twoH + c];
    ctx[(size_t)b * twoH + c] = acc;
  }
}

// x_bf(64,3328) = bf16(concat(cx_e, cx_p, cx_q, w))
__global__ void k_build_x(const float* __restrict__ cxe, const float* __restrict__ cxp,
                          const float* __restrict__ cxq, const float* __restrict__ wref,
                          bf16_t* __restrict__ xbf) {
  int g = blockIdx.x * blockDim.x + threadIdx.x;
  if (g >= 64 * 3328) return;
  int b = g / 3328, c = g % 3328;
  float v;
  if      (c < 1024) v = cxe[b * 1024 + c];
  else if (c < 2048) v = cxp[b * 1024 + (c - 1024)];
  else if (c < 3072) v = cxq[b * 1024 + (c - 2048)];
  else               v = wref[b * 256 + (c - 3072)];
  xbf[g] = (bf16_t)v;
}

// p_gen[b] = sigmoid(cx_e.Wctx + s.Wdec + w.Ww + biases)  (s = updated state)
__global__ void k_pgen(const float* __restrict__ cxe, const float* __restrict__ s,
                       const float* __restrict__ w,
                       const float* WctxW, const float* Wctxb,
                       const float* WdecW, const float* Wdecb,
                       const float* WwW, const float* Wwb,
                       float* __restrict__ pgen) {
  int b = blockIdx.x, tid = threadIdx.x;        // 256
  float acc = 0.0f;
  for (int j = tid; j < 1024; j += 256) acc += cxe[b * 1024 + j] * WctxW[j];
  for (int j = tid; j < 512;  j += 256) acc += s[b * 512 + j] * WdecW[j];
  if (tid < 256) acc += w[b * 256 + tid] * WwW[tid];
  __shared__ float red[256];
  red[tid] = acc;
  for (int st = 128; st > 0; st >>= 1) { __syncthreads(); if (tid < st) red[tid] += red[tid + st]; }
  if (tid == 0) pgen[b] = sigm(red[0] + Wctxb[0] + Wdecb[0] + Wwb[0]);
}

// out_row = log(p_gen) + log_softmax(logits); then copy scatter-add at entities
__global__ __launch_bounds__(1024)
void k_logsoftmax_copy(const float* __restrict__ logits, const float* __restrict__ pgen,
                       const float* __restrict__ ae, const int* __restrict__ entities,
                       float* __restrict__ outp, int B, int V, int LE) {
  int b = blockIdx.x, tid = threadIdx.x;        // 1024
  const float* row = logits + (size_t)b * V;
  float* orow = outp + (size_t)b * V;
  __shared__ float red[1024];
  float m = -1e30f;
  for (int v = tid; v < V; v += 1024) m = fmaxf(m, row[v]);
  red[tid] = m;
  for (int s = 512; s > 0; s >>= 1) { __syncthreads(); if (tid < s) red[tid] = fmaxf(red[tid], red[tid + s]); }
  __syncthreads(); m = red[0]; __syncthreads();
  float sum = 0.0f;
  for (int v = tid; v < V; v += 1024) sum += __expf(row[v] - m);
  red[tid] = sum;
  for (int s = 512; s > 0; s >>= 1) { __syncthreads(); if (tid < s) red[tid] += red[tid + s]; }
  __syncthreads(); sum = red[0];
  float lz = m + logf(sum);
  float pg = pgen[b];
  float lpg = logf(pg);
  for (int v = tid; v < V; v += 1024) orow[v] = lpg + (row[v] - lz);
  __syncthreads();
  if (tid < LE) {
    int vv = entities[tid * B + b];             // entities: (LE,B)
    orow[vv] += logf(1.0f - pg + ae[b * LE + tid] + 1e-7f);
  }
}

// ---------------------------------------------------------------------------
// Host orchestration
// ---------------------------------------------------------------------------
extern "C" void kernel_launch(void* const* d_in, const int* in_sizes, int n_in,
                              void* d_out, int out_size, void* d_ws, size_t ws_size,
                              hipStream_t stream) {
  (void)in_sizes; (void)n_in; (void)out_size; (void)ws_size;
  const int E = 256, H = 512, Adim = 512, V = 32000, LE = 8, LC = 128, T = 30, B = 64;

  // --- input map (setup_inputs insertion order, params flattened recursively)
  const int*   entities = (const int*)d_in[0];
  const int*   pre_ctx  = (const int*)d_in[1];
  const int*   post_ctx = (const int*)d_in[2];
  const int*   refexes  = (const int*)d_in[3];
  const float* emb      = (const float*)d_in[4];
  struct LstmP { const float *Wih, *Whh, *bih, *bhh; };
  auto getL = [&](int i) { return LstmP{ (const float*)d_in[i], (const float*)d_in[i+1],
                                         (const float*)d_in[i+2], (const float*)d_in[i+3] }; };
  LstmP ent_f = getL(5),  ent_b = getL(9),  pre_f = getL(13),
        pre_b = getL(17), post_f = getL(21), post_b = getL(25);
  struct AttP { const float *WencW, *Wencb, *WdecW, *Wdecb, *vW, *vb; };
  auto getA = [&](int i) { return AttP{ (const float*)d_in[i], (const float*)d_in[i+1],
                                        (const float*)d_in[i+2], (const float*)d_in[i+3],
                                        (const float*)d_in[i+4], (const float*)d_in[i+5] }; };
  AttP atts[3] = { getA(29), getA(35), getA(41) };           // ent, pre, post
  const float* cpWctxW = (const float*)d_in[47]; const float* cpWctxb = (const float*)d_in[48];
  const float* cpWdecW = (const float*)d_in[49]; const float* cpWdecb = (const float*)d_in[50];
  const float* cpWwW   = (const float*)d_in[51]; const float* cpWwb   = (const float*)d_in[52];
  LstmP dec = getL(53);
  const float* outW = (const float*)d_in[57];
  const float* outb = (const float*)d_in[58];
  float* out = (float*)d_out;

  // --- workspace bump allocator
  char* wsb = (char*)d_ws;
  size_t off = 0;
  auto alloc  = [&](size_t bytes) { void* p = wsb + off; off = (off + bytes + 255) & ~(size_t)255; return p; };
  auto allocf = [&](size_t n) { return (float*)alloc(n * 4); };
  auto allocb = [&](size_t n) { return (bf16_t*)alloc(n * 2); };

  auto cvt  = [&](const float* s, bf16_t* d, size_t n) {
    k_f32_to_bf16<<<(unsigned)((n + 255) / 256), 256, 0, stream>>>(s, d, (int)n); };
  auto zero = [&](float* p, size_t n) {
    k_zero_f32<<<(unsigned)((n + 255) / 256), 256, 0, stream>>>(p, (int)n); };
  auto gemm = [&](const bf16_t* A, const bf16_t* W, const float* bias, float* C,
                  int M, int N, int K) {
    int grid = (M / 16) * (N / 256);            // block = 16x256 output strip
    size_t shbytes = (size_t)16 * K * 2;        // A tile staged in LDS by TDM
    k_gemm<<<grid, 128, shbytes, stream>>>(A, W, bias, C, M, N, K); };

  // --- encoder runs
  struct Run { LstmP p; const int* idx; int L; int rev;
               bf16_t *Wih_bf, *Whh_bf, *X; float *Gx, *hs; };
  Run runs[6] = {
    { ent_f,  entities, LE, 0, 0,0,0,0,0 }, { ent_b,  entities, LE, 1, 0,0,0,0,0 },
    { pre_f,  pre_ctx,  LC, 0, 0,0,0,0,0 }, { pre_b,  pre_ctx,  LC, 1, 0,0,0,0,0 },
    { post_f, post_ctx, LC, 0, 0,0,0,0,0 }, { post_b, post_ctx, LC, 1, 0,0,0,0,0 } };

  for (int i = 0; i < 6; ++i) {
    Run& r = runs[i];
    r.Wih_bf = allocb((size_t)2048 * E);
    r.Whh_bf = allocb((size_t)2048 * H);
    r.X      = allocb((size_t)r.L * B * E);
    r.Gx     = allocf((size_t)r.L * B * 2048);
    r.hs     = allocf((size_t)r.L * B * H);
    cvt(r.p.Wih, r.Wih_bf, (size_t)2048 * E);
    cvt(r.p.Whh, r.Whh_bf, (size_t)2048 * H);
    int tot = r.L * B * E;
    k_gather_bf16<<<(tot + 255) / 256, 256, 0, stream>>>(emb, r.idx, r.X, r.L, B, E, r.rev);
    gemm(r.X, r.Wih_bf, r.p.bih, r.Gx, r.L * B, 2048, E);   // includes bih
  }

  // all 6 recurrent runs in ONE launch: one persistent workgroup per run
  {
    LstmArgs la{};
    for (int i = 0; i < 6; ++i)
      la.d[i] = LstmDesc{ runs[i].Gx, runs[i].Whh_bf, runs[i].p.bhh, runs[i].hs,
                          nullptr, nullptr, nullptr, nullptr, runs[i].L, 0 };
    k_lstm<<<6, 1024, 2 * 64 * 512 * 2, stream>>>(la);      // 128KB dynamic LDS
  }

  // --- enc concat + attention pre-projections
  int Ls[3] = { LE, LC, LC };
  float  *encF[3], *eenc[3];
  bf16_t *encB[3], *WencBf[3], *WdecBf[3];
  for (int gi = 0; gi < 3; ++gi) {
    int L = Ls[gi];
    encF[gi]   = allocf((size_t)L * B * 2 * H);
    encB[gi]   = allocb((size_t)L * B * 2 * H);
    eenc[gi]   = allocf((size_t)L * B * Adim);
    WencBf[gi] = allocb((size_t)Adim * 2 * H);
    WdecBf[gi] = allocb((size_t)Adim * H);
    cvt(atts[gi].WencW, WencBf[gi], (size_t)Adim * 2 * H);
    cvt(atts[gi].WdecW, WdecBf[gi], (size_t)Adim * H);
    int tot = L * B * 2 * H;
    k_concat_enc<<<(tot + 255) / 256, 256, 0, stream>>>(runs[2*gi].hs, runs[2*gi+1].hs,
                                                        encF[gi], encB[gi], L, B, H);
    gemm(encB[gi], WencBf[gi], atts[gi].Wencb, eenc[gi], L * B, Adim, 2 * H);
  }

  // --- decoder weights / buffers
  bf16_t* decWih_bf = allocb((size_t)2048 * 3328);
  bf16_t* decWhh_bf = allocb((size_t)2048 * H);
  bf16_t* outW_bf   = allocb((size_t)V * H);
  cvt(dec.Wih, decWih_bf, (size_t)2048 * 3328);
  cvt(dec.Whh, decWhh_bf, (size_t)2048 * H);
  cvt(outW,    outW_bf,   (size_t)V * H);

  float* embref = allocf((size_t)T * B * E);
  k_gather_f32<<<(T * B * E + 255) / 256, 256, 0, stream>>>(emb, refexes, embref, T * B, E);

  float*  sbuf = allocf((size_t)B * H);
  float*  cbuf = allocf((size_t)B * H);
  bf16_t* sbf  = allocb((size_t)B * H);
  float*  sdec[3]; float* scoreb[3]; float* ctx[3];
  for (int i = 0; i < 3; ++i) {
    sdec[i]   = allocf((size_t)B * Adim);
    scoreb[i] = allocf((size_t)LC * B);
    ctx[i]    = allocf((size_t)B * 2 * H);
  }
  float*  ae     = allocf((size_t)B * LE);
  bf16_t* xbf    = allocb((size_t)B * 3328);
  float*  Gdec   = allocf((size_t)B * 2048);
  float*  pgen   = allocf((size_t)B);
  float*  logits = allocf((size_t)B * V);

  zero(sbuf, (size_t)B * H);
  zero(cbuf, (size_t)B * H);
  zero((float*)sbf, (size_t)B * H / 2);                     // bf16 zeros

  // --- decoder loop: 29 steps
  for (int t = 0; t < T - 1; ++t) {
    // attention (uses previous state s via sbf)
    for (int i = 0; i < 3; ++i) {
      gemm(sbf, WdecBf[i], atts[i].Wdecb, sdec[i], B, Adim, H);
      k_att_score<<<Ls[i] * B, 128, 0, stream>>>(eenc[i], sdec[i], atts[i].vW, atts[i].vb,
                                                 scoreb[i], B);
      k_softmax_ctx<<<B, 256, 0, stream>>>(scoreb[i], encF[i], ctx[i],
                                           (i == 0) ? ae : nullptr, Ls[i], B, 2 * H);
    }
    // decoder LSTM cell input
    const float* wref = embref + (size_t)t * B * E;
    k_build_x<<<(64 * 3328 + 255) / 256, 256, 0, stream>>>(ctx[0], ctx[1], ctx[2], wref, xbf);
    gemm(xbf, decWih_bf, dec.bih, Gdec, B, 2048, 3328);
    {
      LstmArgs ld{};
      ld.d[0] = LstmDesc{ Gdec, decWhh_bf, dec.bhh, sbuf, sbuf, cbuf, cbuf, sbf, 1, 0 };
      k_lstm<<<1, 1024, 2 * 64 * 512 * 2, stream>>>(ld);
    }
    // vocab projection + copy mechanism (p_gen uses the UPDATED state)
    gemm(sbf, outW_bf, outb, logits, B, V, H);
    k_pgen<<<B, 256, 0, stream>>>(ctx[0], sbuf, wref, cpWctxW, cpWctxb,
                                  cpWdecW, cpWdecb, cpWwW, cpWwb, pgen);
    k_logsoftmax_copy<<<B, 1024, 0, stream>>>(logits, pgen, ae, entities,
                                              out + (size_t)t * B * V, B, V, LE);
  }
}
